// BPPSLodeModel_78073915506942
// MI455X (gfx1250) — matrix operand order repaired and checked by reading, not compiled, not run on hardware
//
#include <hip/hip_runtime.h>

// ---------------------------------------------------------------------------
// MI455X (gfx1250) fused BP-species-MLP, round 2.
//  - bf16 WMMA (v_wmma_f32_16x16x32_bf16), f32 accumulation
//  - per wave: 2 M-tiles x 8 N-tiles -> every B-fragment (global_load_b128 x2)
//    feeds two WMMAs; the block's two waves split N, so each weight matrix is
//    streamed from L2 exactly once per 32-atom block (2x less than round 1)
//  - single-HBM-pass LayerNorm: raw x stashed to LDS as bf16 during the
//    statistics pass, normalized in-LDS (x read from HBM exactly once)
//  - weights pre-transposed to [S][H][K] bf16 in d_ws (prep kernel)
//  - deterministic per-molecule reduction (no float atomics)
// ---------------------------------------------------------------------------

typedef __attribute__((ext_vector_type(16))) __bf16 v16bf;
typedef __attribute__((ext_vector_type(8)))  __bf16 v8bf;
typedef __attribute__((ext_vector_type(8)))  float  v8f;

union FragAB { v16bf v; v8bf h[2]; };

#define N_ATOMS   65536
#define N_SPECIES 4
#define N_MOL     512
#define HDIM      256
#define HP        264   // padded LDS row stride for H: 132 dwords -> +4 banks/row

__device__ __forceinline__ float silu_f(float v) {
    return v / (1.0f + __expf(-v));
}

__device__ __forceinline__ void zero_acc(v8f* acc) {
#pragma unroll
    for (int i = 0; i < 16; ++i) {
        v8f z = {0.f, 0.f, 0.f, 0.f, 0.f, 0.f, 0.f, 0.f};
        acc[i] = z;
    }
}

// One wave computes a [32 x 128] tile as 2 M-tiles x 8 N-tiles.
// A rows from LDS (bf16, padded stride); B from transposed bf16 weights,
// rows pre-offset to this wave's N half. acc[mt*8 + nt].
template <int KDIM, int ASTRIDE>
__device__ __forceinline__ void wave_gemm_2x8(const __bf16* __restrict__ aRows,
                                              const __bf16* __restrict__ wTw,
                                              v8f* acc, int lane) {
    const int m    = lane & 15;   // A row in tile == B column in n-tile
    const int half = lane >> 4;
    const __bf16* a0row = aRows + (size_t)m * ASTRIDE;
    const __bf16* a1row = aRows + (size_t)(16 + m) * ASTRIDE;
    for (int k0 = 0; k0 < KDIM; k0 += 32) {
        FragAB a0, a1;
        const __bf16* ap0 = a0row + k0 + 8 * half;
        const __bf16* ap1 = a1row + k0 + 8 * half;
        a0.h[0] = *reinterpret_cast<const v8bf*>(ap0);        // K = 8h..8h+7
        a0.h[1] = *reinterpret_cast<const v8bf*>(ap0 + 16);   // K = 16+8h..
        a1.h[0] = *reinterpret_cast<const v8bf*>(ap1);
        a1.h[1] = *reinterpret_cast<const v8bf*>(ap1 + 16);
#pragma unroll
        for (int nt = 0; nt < 8; ++nt) {
            FragAB b;
            const __bf16* bp = wTw + (size_t)(nt * 16 + m) * KDIM + k0 + 8 * half;
            b.h[0] = *reinterpret_cast<const v8bf*>(bp);      // K = 8h..8h+7
            b.h[1] = *reinterpret_cast<const v8bf*>(bp + 16); // K = 16+8h..
            acc[nt] = __builtin_amdgcn_wmma_f32_16x16x32_bf16(
                false, a0.v, false, b.v, (short)0, acc[nt], false, false);
            acc[8 + nt] = __builtin_amdgcn_wmma_f32_16x16x32_bf16(
                false, a1.v, false, b.v, (short)0, acc[8 + nt], false, false);
        }
    }
}

// SiLU on f32 accumulators, store bf16 into LDS rows (stride dstride).
// Wave covers rows 0..31, columns nbase..nbase+127.
__device__ __forceinline__ void silu_store_2x8(const v8f* acc, __bf16* dst,
                                               int dstride, int lane, int nbase) {
    const int n    = lane & 15;
    const int half = lane >> 4;
#pragma unroll
    for (int mt = 0; mt < 2; ++mt) {
#pragma unroll
        for (int nt = 0; nt < 8; ++nt) {
#pragma unroll
            for (int r = 0; r < 8; ++r) {
                const int row = mt * 16 + r + 8 * half;  // C/D: M = vgpr + 8*(lane/16)
                dst[(size_t)row * dstride + nbase + nt * 16 + n] =
                    (__bf16)silu_f(acc[mt * 8 + nt][r]);
            }
        }
    }
}

// Fused LayerNorm + species MLP. Block = 64 threads (2 waves), 32 atoms/block.
// ACCUM=0: e_out[atom] = e;  ACCUM=1: e_out[atom] += e.
template <int F, int ACCUM>
__global__ __launch_bounds__(64) void bp_species_mlp(
    const float* __restrict__ x, const float* __restrict__ gamma,
    const float* __restrict__ beta, const __bf16* __restrict__ W1T,
    const __bf16* __restrict__ W2T, const float* __restrict__ W3,
    float* __restrict__ e_out) {
    constexpr int XP  = F + 8;   // padded input row stride
    constexpr int NCH = F / 32;  // feature chunks per lane
    __shared__ __align__(16) __bf16 xn[32 * XP];  // LN output; reused as h2 [32][HP]
    __shared__ __align__(16) __bf16 h1[32 * HP];

    const int lane     = threadIdx.x & 31;
    const int wid      = threadIdx.x >> 5;
    const int atomBase = blockIdx.x * 32;
    const int s        = atomBase >> 14;  // 16384 atoms/species, block-aligned
    const int nbase    = wid * 128;       // this wave's N half

    // ---- LayerNorm: wave handles its own 16 rows; x read from HBM once ----
    {
        float g[NCH], bt[NCH];
#pragma unroll
        for (int j = 0; j < NCH; ++j) {
            g[j]  = gamma[lane + 32 * j];
            bt[j] = beta[lane + 32 * j];
        }
        for (int lr = 0; lr < 16; ++lr) {
            const int row   = wid * 16 + lr;
            const float* xr = x + (size_t)(atomBase + row) * F;
            __bf16* xrow    = &xn[(size_t)row * XP];
            float sum = 0.f, sq = 0.f;
#pragma unroll
            for (int j = 0; j < NCH; ++j) {
                const float v = xr[lane + 32 * j];
                sum += v;
                sq  += v * v;
                xrow[lane + 32 * j] = (__bf16)v;  // stash raw as bf16
            }
#pragma unroll
            for (int o = 16; o > 0; o >>= 1) {
                sum += __shfl_xor(sum, o, 32);
                sq  += __shfl_xor(sq,  o, 32);
            }
            const float mu  = sum * (1.0f / F);
            const float var = sq * (1.0f / F) - mu * mu;
            const float rs  = rsqrtf(var + 1e-5f);
#pragma unroll
            for (int j = 0; j < NCH; ++j) {
                const float v = (float)xrow[lane + 32 * j];
                xrow[lane + 32 * j] = (__bf16)((v - mu) * rs * g[j] + bt[j]);
            }
        }
    }
    __syncthreads();

    // ---- Layer 1: [32,F] x [F,128-half] -> SiLU -> h1 ----
    v8f acc[16];
    zero_acc(acc);
    wave_gemm_2x8<F, XP>(xn, W1T + ((size_t)s * HDIM + nbase) * F, acc, lane);
    silu_store_2x8(acc, h1, HP, lane, nbase);
    __syncthreads();

    // ---- Layer 2: [32,256] x [256,128-half] -> SiLU -> h2 (reuses xn) ----
    zero_acc(acc);
    __bf16* h2 = xn;
    wave_gemm_2x8<HDIM, HP>(h1, W2T + ((size_t)s * HDIM + nbase) * HDIM, acc, lane);
    silu_store_2x8(acc, h2, HP, lane, nbase);
    __syncthreads();

    // ---- Layer 3: e[row] = dot(h2[row], W3[s]) ; 2 lanes per row ----
    const int lr   = lane >> 1;
    const int part = lane & 1;
    const __bf16* hrow = &h2[(size_t)(wid * 16 + lr) * HP + part * 128];
    const float*  w3   = W3 + s * HDIM + part * 128;
    float e = 0.f;
    for (int j = 0; j < 128; ++j) e += (float)hrow[j] * w3[j];
    e += __shfl_xor(e, 1, 32);
    if (part == 0) {
        const int atom = atomBase + wid * 16 + lr;
        if (ACCUM)
            e_out[atom] += e;
        else
            e_out[atom] = e;
    }
}

// Convert W [S][F][H] f32 -> WT [S][H][F] bf16 (WMMA B-fragments become
// contiguous 16B chunks per lane).
__global__ __launch_bounds__(256) void prep_wT(const float* __restrict__ W,
                                               __bf16* __restrict__ WT, int F, int H) {
    const int idx   = blockIdx.x * 256 + threadIdx.x;
    const int total = N_SPECIES * F * H;
    if (idx >= total) return;
    const int s   = idx / (F * H);
    const int rem = idx - s * F * H;
    const int f   = rem / H;
    const int h   = rem - f * H;
    WT[(size_t)s * F * H + (size_t)h * F + f] = (__bf16)W[idx];
}

// Deterministic segment_sum: one block per molecule scans all atoms
// (batch[] and e[] are L2-resident: 512 KB total).
__global__ __launch_bounds__(256) void reduce_mol(const float* __restrict__ e,
                                                  const int* __restrict__ batch,
                                                  float* __restrict__ out) {
    __shared__ float red[256];
    const int m = blockIdx.x;
    float s = 0.f;
    for (int i = threadIdx.x; i < N_ATOMS; i += 256)
        if (batch[i] == m) s += e[i];
    red[threadIdx.x] = s;
    __syncthreads();
    for (int w = 128; w > 0; w >>= 1) {
        if ((int)threadIdx.x < w) red[threadIdx.x] += red[threadIdx.x + w];
        __syncthreads();
    }
    if (threadIdx.x == 0) out[m] = red[0];
}

extern "C" void kernel_launch(void* const* d_in, const int* in_sizes, int n_in,
                              void* d_out, int out_size, void* d_ws, size_t ws_size,
                              hipStream_t stream) {
    (void)in_sizes; (void)n_in; (void)out_size; (void)ws_size;

    const float* x_ps     = (const float*)d_in[0];
    const float* x_mp     = (const float*)d_in[1];
    const int*   batch    = (const int*)d_in[2];
    const float* gamma_ps = (const float*)d_in[3];
    const float* beta_ps  = (const float*)d_in[4];
    const float* gamma_mp = (const float*)d_in[5];
    const float* beta_mp  = (const float*)d_in[6];
    const float* W_ps1    = (const float*)d_in[7];
    const float* W_ps2    = (const float*)d_in[8];
    const float* W_ps3    = (const float*)d_in[9];
    const float* W_mp1    = (const float*)d_in[10];
    const float* W_mp2    = (const float*)d_in[11];
    const float* W_mp3    = (const float*)d_in[12];
    float* out = (float*)d_out;

    // Workspace layout (bytes):
    //   W1T_ps 4*256*512*2 = 1048576
    //   W2T_ps 4*256*256*2 =  524288
    //   W1T_mp 4*256*256*2 =  524288
    //   W2T_mp 4*256*256*2 =  524288
    //   e_atoms 65536*4    =  262144   (total ~2.9 MB)
    char* ws = (char*)d_ws;
    __bf16* W1T_ps  = (__bf16*)(ws);
    __bf16* W2T_ps  = (__bf16*)(ws + 1048576);
    __bf16* W1T_mp  = (__bf16*)(ws + 1048576 + 524288);
    __bf16* W2T_mp  = (__bf16*)(ws + 1048576 + 2 * 524288);
    float*  e_atoms = (float*)(ws + 1048576 + 3 * 524288);

    // Stage bf16 transposed weights.
    prep_wT<<<(N_SPECIES * 512 * 256 + 255) / 256, 256, 0, stream>>>(W_ps1, W1T_ps, 512, 256);
    prep_wT<<<(N_SPECIES * 256 * 256 + 255) / 256, 256, 0, stream>>>(W_ps2, W2T_ps, 256, 256);
    prep_wT<<<(N_SPECIES * 256 * 256 + 255) / 256, 256, 0, stream>>>(W_mp1, W1T_mp, 256, 256);
    prep_wT<<<(N_SPECIES * 256 * 256 + 255) / 256, 256, 0, stream>>>(W_mp2, W2T_mp, 256, 256);

    // Fused LN + MLP, both paths (same stream => ordered; mp accumulates).
    bp_species_mlp<512, 0><<<N_ATOMS / 32, 64, 0, stream>>>(
        x_ps, gamma_ps, beta_ps, W1T_ps, W2T_ps, W_ps3, e_atoms);
    bp_species_mlp<256, 1><<<N_ATOMS / 32, 64, 0, stream>>>(
        x_mp, gamma_mp, beta_mp, W1T_mp, W2T_mp, W_mp3, e_atoms);

    // Deterministic per-molecule reduction.
    reduce_mol<<<N_MOL, 256, 0, stream>>>(e_atoms, batch, out);
}
